// SimpleGGNN_22325240004844
// MI455X (gfx1250) — compile-verified, matrix operand order checked by reading, not run on hardware
//
#include <hip/hip_runtime.h>
#include <hip/hip_bf16.h>

#define HID 128
#define NT 8

typedef __attribute__((ext_vector_type(16))) __bf16        v16bf;
typedef __attribute__((ext_vector_type(8)))  float         v8f;
typedef __attribute__((ext_vector_type(4)))  float         v4f;
typedef __attribute__((ext_vector_type(4)))  unsigned int  v4u;
typedef __attribute__((ext_vector_type(2)))  unsigned int  v2u;

struct bfrag {
    union {
        v16bf          v;
        unsigned short u[16];
        v4u            q[2];
    };
};

// Native f32 -> bf16 (RNE): backend selects v_cvt_(pk_)bf16_f32 on gfx1250.
__device__ __forceinline__ unsigned short f2bf(float f) {
    __bf16 b = (__bf16)f;
    return __builtin_bit_cast(unsigned short, b);
}
__device__ __forceinline__ float bf2f(unsigned short s) {
    return __uint_as_float(((unsigned int)s) << 16);
}

// Build a 16x32 bf16 A-fragment from an fp32 row.
// Lane l: M = l&15; K = k0..k0+7 (frag elems 0-7) and k0+16..k0+23 (elems 8-15),
// where k0 = ks*32 + (l>=16 ? 8 : 0).
__device__ __forceinline__ void load_a_frag_f32(bfrag& f, const float* __restrict__ row, int k0) {
    v4f x0 = *(const v4f*)(row + k0);
    v4f x1 = *(const v4f*)(row + k0 + 4);
    v4f x2 = *(const v4f*)(row + k0 + 16);
    v4f x3 = *(const v4f*)(row + k0 + 20);
#pragma unroll
    for (int j = 0; j < 4; ++j) {
        f.v[j]      = (__bf16)x0[j];
        f.v[4 + j]  = (__bf16)x1[j];
        f.v[8 + j]  = (__bf16)x2[j];
        f.v[12 + j] = (__bf16)x3[j];
    }
}

// ---------------- prep kernels ----------------

// Wt[t][n][k] = W[t][k][n] as bf16  (so B-fragment loads are contiguous per lane)
__global__ void ggnn_cvt_transpose_W(const float* __restrict__ W, unsigned short* __restrict__ Wt) {
    int idx = blockIdx.x * blockDim.x + threadIdx.x;
    if (idx >= NT * HID * HID) return;
    int k = idx % HID;
    int n = (idx / HID) % HID;
    int t = idx / (HID * HID);
    Wt[idx] = f2bf(W[(size_t)t * HID * HID + (size_t)k * HID + n]);
}

__global__ void ggnn_cvt_bf16(const float* __restrict__ src, unsigned short* __restrict__ dst, int n) {
    int idx = blockIdx.x * blockDim.x + threadIdx.x;
    if (idx < n) dst[idx] = f2bf(src[idx]);
}

// ---------------- kernel B: all_msgs[n,t,:] = h[n] @ W[t] + b[t]  (bf16 out) ----------------

__global__ __launch_bounds__(256) void ggnn_msgs_gemm(
        const float* __restrict__ h,            // (N,128) fp32
        const unsigned short* __restrict__ Wt,  // (T,128out,128in) bf16
        const float* __restrict__ bias,         // (T,128) fp32
        unsigned short* __restrict__ all_msgs,  // (N,T,128) bf16
        int n_nodes) {
    int tile = blockIdx.x;
    int wave = threadIdx.x >> 5;     // = edge type t
    int lane = threadIdx.x & 31;
    int half = lane >> 4;
    int l16  = lane & 15;
    int row0 = tile * 16;
    int t    = wave;
    bool full = (row0 + 16) <= n_nodes;   // uniform across block

    int rr = row0 + l16;
    if (rr >= n_nodes) rr = n_nodes - 1;
    const float* hrow = h + (size_t)rr * HID;

    bfrag a[4];
#pragma unroll
    for (int ks = 0; ks < 4; ++ks) load_a_frag_f32(a[ks], hrow, ks * 32 + half * 8);

#pragma unroll
    for (int nc = 0; nc < 8; ++nc) {
        const unsigned short* wr = Wt + ((size_t)t * HID + (size_t)(nc * 16 + l16)) * HID;
        v8f acc = {};
#pragma unroll
        for (int ks = 0; ks < 4; ++ks) {
            bfrag bb;
            int k0 = ks * 32 + half * 8;
            bb.q[0] = *(const v4u*)(wr + k0);
            bb.q[1] = *(const v4u*)(wr + k0 + 16);
            acc = __builtin_amdgcn_wmma_f32_16x16x32_bf16(
                    false, a[ks].v, false, bb.v, (short)0, acc, false, false);
        }
        float bv = bias[t * HID + nc * 16 + l16];
#pragma unroll
        for (int i = 0; i < 8; ++i) acc[i] += bv;

        unsigned short* obase = all_msgs + (size_t)(row0 + 8 * half) * NT * HID
                                         + (size_t)t * HID + nc * 16 + l16;
        if (full) {
            // branch-free fast path: row stride NT*HID elements
#pragma unroll
            for (int r = 0; r < 8; ++r)
                obase[(size_t)r * NT * HID] = f2bf(acc[r]);
        } else {
#pragma unroll
            for (int r = 0; r < 8; ++r) {
                int m = row0 + r + 8 * half;
                if (m < n_nodes)
                    all_msgs[((size_t)m * NT + t) * HID + nc * 16 + l16] = f2bf(acc[r]);
            }
        }
    }
}

// ---------------- kernel C: scatter-add gathered messages into dst nodes ----------------

__global__ __launch_bounds__(256) void ggnn_scatter(
        const int* __restrict__ edge_index,     // (2,E)
        const int* __restrict__ edge_type,      // (E)
        const unsigned short* __restrict__ all_msgs,
        float* __restrict__ messages,           // (N,128) fp32, pre-zeroed
        int n_edges) {
    int e    = (int)((blockIdx.x * (unsigned)blockDim.x + threadIdx.x) >> 5);
    int lane = threadIdx.x & 31;
    if (e >= n_edges) return;
    int src = edge_index[e];
    int dst = edge_index[n_edges + e];
    int t   = edge_type[e];
    const unsigned short* m = all_msgs + ((size_t)src * NT + (size_t)t) * HID;
    float* out = messages + (size_t)dst * HID;
    v2u raw = *(const v2u*)(m + lane * 4);      // 4 bf16 per lane
    atomicAdd(out + lane * 4 + 0, bf2f((unsigned short)(raw.x & 0xFFFFu)));
    atomicAdd(out + lane * 4 + 1, bf2f((unsigned short)(raw.x >> 16)));
    atomicAdd(out + lane * 4 + 2, bf2f((unsigned short)(raw.y & 0xFFFFu)));
    atomicAdd(out + lane * 4 + 3, bf2f((unsigned short)(raw.y >> 16)));
}

// ---------------- kernel D: fused GRU (two GEMMs + gates) ----------------

__global__ __launch_bounds__(256) void ggnn_gru(
        const float* __restrict__ h,            // (N,128) fp32
        const float* __restrict__ messages,     // (N,128) fp32
        const unsigned short* __restrict__ wih, // (384,128) bf16
        const unsigned short* __restrict__ whh, // (384,128) bf16
        const float* __restrict__ bih,          // (384)
        const float* __restrict__ bhh,          // (384)
        float* __restrict__ out,                // (N,128) fp32
        int n_nodes) {
    int tile = blockIdx.x;
    int c    = threadIdx.x >> 5;   // output column tile 0..7
    int lane = threadIdx.x & 31;
    int half = lane >> 4;
    int l16  = lane & 15;
    int row0 = tile * 16;
    bool full = (row0 + 16) <= n_nodes;

    int rr = row0 + l16;
    if (rr >= n_nodes) rr = n_nodes - 1;
    const float* mrow = messages + (size_t)rr * HID;
    const float* hrow = h        + (size_t)rr * HID;

    bfrag am[4], ah[4];
#pragma unroll
    for (int ks = 0; ks < 4; ++ks) {
        int k0 = ks * 32 + half * 8;
        load_a_frag_f32(am[ks], mrow, k0);
        load_a_frag_f32(ah[ks], hrow, k0);
    }

    v8f g[6];   // i_r, i_z, i_n, h_r, h_z, h_n
#pragma unroll
    for (int gt = 0; gt < 6; ++gt) {
        const unsigned short* wmat = (gt < 3) ? wih : whh;
        const bfrag*          afr  = (gt < 3) ? am  : ah;
        int grow = (gt % 3) * HID + c * 16 + l16;   // weight-matrix row = output index
        const unsigned short* wr = wmat + (size_t)grow * HID;
        v8f acc = {};
#pragma unroll
        for (int ks = 0; ks < 4; ++ks) {
            bfrag bb;
            int k0 = ks * 32 + half * 8;
            bb.q[0] = *(const v4u*)(wr + k0);
            bb.q[1] = *(const v4u*)(wr + k0 + 16);
            acc = __builtin_amdgcn_wmma_f32_16x16x32_bf16(
                    false, afr[ks].v, false, bb.v, (short)0, acc, false, false);
        }
        float bv = (gt < 3) ? bih[(gt % 3) * HID + c * 16 + l16]
                            : bhh[(gt % 3) * HID + c * 16 + l16];
#pragma unroll
        for (int i = 0; i < 8; ++i) acc[i] += bv;
        g[gt] = acc;
    }

    int col = c * 16 + l16;
    const float* hbase = h   + (size_t)(row0 + 8 * half) * HID + col;
    float*       obase = out + (size_t)(row0 + 8 * half) * HID + col;
#pragma unroll
    for (int r = 0; r < 8; ++r) {
        int m = row0 + r + 8 * half;
        if (!full && m >= n_nodes) continue;
        float i_r = g[0][r], i_z = g[1][r], i_n = g[2][r];
        float h_r = g[3][r], h_z = g[4][r], h_n = g[5][r];
        float rg = 1.0f / (1.0f + __expf(-(i_r + h_r)));
        float zg = 1.0f / (1.0f + __expf(-(i_z + h_z)));
        float ng = tanhf(i_n + rg * h_n);
        float hv = hbase[(size_t)r * HID];
        obase[(size_t)r * HID] = (1.0f - zg) * ng + zg * hv;
    }
}

// ---------------- host launcher ----------------

extern "C" void kernel_launch(void* const* d_in, const int* in_sizes, int n_in,
                              void* d_out, int out_size, void* d_ws, size_t ws_size,
                              hipStream_t stream) {
    const float* h        = (const float*)d_in[0];
    const int*   edge_idx = (const int*)d_in[1];
    const int*   edge_typ = (const int*)d_in[2];
    const float* W        = (const float*)d_in[3];
    const float* bias     = (const float*)d_in[4];
    const float* w_ih     = (const float*)d_in[5];
    const float* w_hh     = (const float*)d_in[6];
    const float* b_ih     = (const float*)d_in[7];
    const float* b_hh     = (const float*)d_in[8];
    float* out = (float*)d_out;

    int n_nodes = in_sizes[0] / HID;     // 100000
    int n_edges = in_sizes[2];           // 600000

    // workspace carve-up (all chunks 256B-aligned by construction)
    char* p = (char*)d_ws;
    unsigned short* Wt   = (unsigned short*)p;  p += (size_t)NT * HID * HID * 2;
    unsigned short* wihb = (unsigned short*)p;  p += (size_t)3 * HID * HID * 2;
    unsigned short* whhb = (unsigned short*)p;  p += (size_t)3 * HID * HID * 2;
    float* messages      = (float*)p;           p += (size_t)n_nodes * HID * 4;
    unsigned short* amsg = (unsigned short*)p;  // (N, T, H) bf16

    // prep
    {
        int n = NT * HID * HID;
        ggnn_cvt_transpose_W<<<(n + 255) / 256, 256, 0, stream>>>(W, Wt);
        int m = 3 * HID * HID;
        ggnn_cvt_bf16<<<(m + 255) / 256, 256, 0, stream>>>(w_ih, wihb, m);
        ggnn_cvt_bf16<<<(m + 255) / 256, 256, 0, stream>>>(w_hh, whhb, m);
    }
    hipMemsetAsync(messages, 0, (size_t)n_nodes * HID * sizeof(float), stream);

    int n_tiles = (n_nodes + 15) / 16;
    ggnn_msgs_gemm<<<n_tiles, 256, 0, stream>>>(h, Wt, bias, amsg, n_nodes);

    int scatter_blocks = (n_edges + 7) / 8;   // 8 waves (=edges) per 256-thread block
    ggnn_scatter<<<scatter_blocks, 256, 0, stream>>>(edge_idx, edge_typ, amsg, messages, n_edges);

    ggnn_gru<<<n_tiles, 256, 0, stream>>>(h, messages, wihb, whhb, b_ih, b_hh, out, n_nodes);
}
